// NSVQ_36567351558900
// MI455X (gfx1250) — matrix operand order repaired and checked by reading, not compiled, not run on hardware
//
#include <hip/hip_runtime.h>
#include <hip/hip_bf16.h>
#include <math.h>

// ---------------- problem constants (from the reference) ----------------
#define N_TOKENS   32768
#define EMB_DIM    256
#define NUM_EMB    8192
#define NSVQ_EPS   1e-8f

// ---------------- CDNA5 WMMA fragment types -----------------------------
typedef __attribute__((ext_vector_type(16))) __bf16 v16bf;
typedef __attribute__((ext_vector_type(8)))  __bf16 v8bf;
typedef __attribute__((ext_vector_type(4)))  __bf16 v4bf;
typedef __attribute__((ext_vector_type(8)))  float  v8f;

union FragBF16 {
    v16bf v;
    v8bf  h[2];
};

struct BFPair { __bf16 hi, lo; };

// Error-compensated bf16 split: x ~= hi + lo (hi,lo both bf16, RNE).
__device__ __forceinline__ BFPair split_bf(float x) {
    BFPair p;
    p.hi = (__bf16)x;
    p.lo = (__bf16)(x - (float)p.hi);
    return p;
}

// ========================================================================
// Kernel 0: codebook squared norms + zero the usage counters.
// Must run every launch (counts accumulate via atomics in the argmin pass).
// ========================================================================
__global__ __launch_bounds__(256)
void nsvq_prep(const float* __restrict__ cb,
               float* __restrict__ cnorm,
               unsigned* __restrict__ counts)
{
    const int k = blockIdx.x * blockDim.x + threadIdx.x;
    if (k >= NUM_EMB) return;
    const float4* row = (const float4*)(cb + (size_t)k * EMB_DIM);
    float s = 0.0f;
#pragma unroll 8
    for (int i = 0; i < EMB_DIM / 4; ++i) {
        float4 v = row[i];
        s += v.x * v.x + v.y * v.y + v.z * v.z + v.w * v.w;
    }
    cnorm[k]  = s;
    counts[k] = 0u;
}

// ========================================================================
// Kernel 1: fused distance GEMM + argmin.  bf16x3 WMMA (hi*hi + hi*lo +
// lo*hi) into f32 accumulators; never materializes the (N,K) matrix.
//
// Block = 256 threads = 8 waves; each wave owns 16 tokens (BM=128/block).
// Codebook staged in 64-code tiles as interleaved bf16 hi/lo rows in a
// double-buffered LDS tile; global loads for tile t+1 are issued before
// the compute of tile t (latency hidden behind 96 WMMAs, 2 barriers/tile).
// Subtiles processed in PAIRS with two interleaved accumulator chains so
// consecutive WMMAs are independent (kills the TRANS32_DEP_1 bubbles).
// Argmin kept per-lane; single 16-lane butterfly after the tile loop.
// ========================================================================
#define BM          128
#define BN          64
#define LO_OFF      264                  // lo row offset within a code block (elems)
#define CODE_STRIDE 536                  // 2*264+8: 268 DWORDs -> conflict-free
#define BUF_ELEMS   (BN * CODE_STRIDE)   // one LDS buffer (34304 elems = 68608 B)

__global__ __launch_bounds__(256)
void nsvq_argmin(const float* __restrict__ X,
                 const float* __restrict__ CB,
                 const float* __restrict__ cnorm,
                 int* __restrict__ minidx,
                 unsigned* __restrict__ counts)
{
    __shared__ __align__(16) __bf16 sData[2 * BUF_ELEMS];   // ~134 KB (<320 KB WGP)

    const int tid  = threadIdx.x;
    const int wave = tid >> 5;
    const int lane = tid & 31;
    const int hv   = lane >> 4;   // half-wave selector
    const int col  = lane & 15;

    const int tokBase = blockIdx.x * BM + wave * 16;
    const int myTok   = tokBase + col;   // A-matrix row M = col (both halves)

    // ---- preload A fragments for all 8 k-steps (contiguous K order, matching B:
    //      lanes 0-15: K 0-15; lanes 16-31: K 16-31) ----
    FragBF16 aHi[8], aLo[8];
    const float* arow = X + (size_t)myTok * EMB_DIM;
#pragma unroll
    for (int ks = 0; ks < 8; ++ks) {
        const int kb = ks * 32;
#pragma unroll
        for (int run = 0; run < 2; ++run) {
            const float4* p = (const float4*)(arow + kb + hv * 16 + run * 8);
            float4 u0 = p[0];
            float4 u1 = p[1];
            float vals[8] = {u0.x, u0.y, u0.z, u0.w, u1.x, u1.y, u1.z, u1.w};
#pragma unroll
            for (int e = 0; e < 8; ++e) {
                BFPair s = split_bf(vals[e]);
                aHi[ks].v[run * 8 + e] = s.hi;
                aLo[ks].v[run * 8 + e] = s.lo;
            }
        }
    }

    // Per-lane running argmin (this lane's column subset); reduced once at end.
    float bestD[8];
    int   bestI[8];
#pragma unroll
    for (int r = 0; r < 8; ++r) { bestD[r] = 3.402823466e38f; bestI[r] = 0; }

    const int numTiles = NUM_EMB / BN;   // 128

    // register-stage tile 0 (16 float4 per thread = one 64-code tile / block)
    float4 stage[16];
    {
        const float4* src = (const float4*)CB;
#pragma unroll
        for (int i = 0; i < 16; ++i) stage[i] = src[tid + i * 256];
    }

#pragma unroll 1
    for (int t = 0; t < numTiles; ++t) {
        __bf16* base = sData + (t & 1) * BUF_ELEMS;

        // ---- convert staged registers -> bf16 hi/lo, store to LDS ----
#pragma unroll
        for (int i = 0; i < 16; ++i) {
            const int lin = tid + i * 256;       // float4 index within tile
            const int row = lin >> 6;            // 64 float4 per code row
            const int c4  = lin & 63;
            float4 u = stage[i];
            v4bf h, l;
            BFPair s0 = split_bf(u.x); h[0] = s0.hi; l[0] = s0.lo;
            BFPair s1 = split_bf(u.y); h[1] = s1.hi; l[1] = s1.lo;
            BFPair s2 = split_bf(u.z); h[2] = s2.hi; l[2] = s2.lo;
            BFPair s3 = split_bf(u.w); h[3] = s3.hi; l[3] = s3.lo;
            *(v4bf*)&base[row * CODE_STRIDE + c4 * 4]          = h;
            *(v4bf*)&base[row * CODE_STRIDE + LO_OFF + c4 * 4] = l;
        }
        __syncthreads();

        // ---- issue global loads for tile t+1 (wait lands next iteration) ----
        if (t + 1 < numTiles) {
            const float4* src = (const float4*)(CB + (size_t)(t + 1) * BN * EMB_DIM);
#pragma unroll
            for (int i = 0; i < 16; ++i) stage[i] = src[tid + i * 256];
        }
        // keep tile t+2 warming in cache
        if (t + 2 < numTiles) {
            __builtin_prefetch(CB + (size_t)(t + 2) * BN * EMB_DIM + tid * 64, 0, 1);
        }

        // ---- 2 pairs of 16-code sub-tiles; two interleaved WMMA chains ----
#pragma unroll
        for (int stp = 0; stp < 2; ++stp) {
            const int c0 = stp * 32 + col;        // sub-tile 2*stp
            const int c1 = stp * 32 + 16 + col;   // sub-tile 2*stp + 1
            const int g0 = t * BN + c0;
            const int g1 = t * BN + c1;
            const float cn0 = cnorm[g0];
            const float cn1 = cnorm[g1];
            const __bf16* b0 = base + c0 * CODE_STRIDE + hv * 16;
            const __bf16* b1 = base + c1 * CODE_STRIDE + hv * 16;

            v8f acc0 = {};
            v8f acc1 = {};
#pragma unroll
            for (int ks = 0; ks < 8; ++ks) {
                // B in this code's contiguous K order (as A): lanes 0-15 K 0-15, lanes 16-31 K 16-31,
                // column N = lane&15 -> contiguous 16 bf16 from the LDS row.
                FragBF16 bHi0, bLo0, bHi1, bLo1;
                bHi0.h[0] = *(const v8bf*)(b0 + ks * 32);
                bHi0.h[1] = *(const v8bf*)(b0 + ks * 32 + 8);
                bLo0.h[0] = *(const v8bf*)(b0 + LO_OFF + ks * 32);
                bLo0.h[1] = *(const v8bf*)(b0 + LO_OFF + ks * 32 + 8);
                bHi1.h[0] = *(const v8bf*)(b1 + ks * 32);
                bHi1.h[1] = *(const v8bf*)(b1 + ks * 32 + 8);
                bLo1.h[0] = *(const v8bf*)(b1 + LO_OFF + ks * 32);
                bLo1.h[1] = *(const v8bf*)(b1 + LO_OFF + ks * 32 + 8);

                // alternate the two chains: consecutive WMMAs are independent
                acc0 = __builtin_amdgcn_wmma_f32_16x16x32_bf16(
                           false, aHi[ks].v, false, bHi0.v, (short)0, acc0, false, false);
                acc1 = __builtin_amdgcn_wmma_f32_16x16x32_bf16(
                           false, aHi[ks].v, false, bHi1.v, (short)0, acc1, false, false);
                acc0 = __builtin_amdgcn_wmma_f32_16x16x32_bf16(
                           false, aHi[ks].v, false, bLo0.v, (short)0, acc0, false, false);
                acc1 = __builtin_amdgcn_wmma_f32_16x16x32_bf16(
                           false, aHi[ks].v, false, bLo1.v, (short)0, acc1, false, false);
                acc0 = __builtin_amdgcn_wmma_f32_16x16x32_bf16(
                           false, aLo[ks].v, false, bHi0.v, (short)0, acc0, false, false);
                acc1 = __builtin_amdgcn_wmma_f32_16x16x32_bf16(
                           false, aLo[ks].v, false, bHi1.v, (short)0, acc1, false, false);
            }

            // dist = ||c||^2 - 2 x.c ; per-lane running min (codes ascend, so
            // strict < preserves lowest-index-on-tie semantics).
#pragma unroll
            for (int r = 0; r < 8; ++r) {
                const float d0 = cn0 - 2.0f * acc0[r];
                if (d0 < bestD[r]) { bestD[r] = d0; bestI[r] = g0; }
                const float d1 = cn1 - 2.0f * acc1[r];
                if (d1 < bestD[r]) { bestD[r] = d1; bestI[r] = g1; }
            }
        }
        __syncthreads();
    }

    // ---- single cross-lane argmin butterfly over the 16 columns/half ----
    // C layout: VGPR r -> row M = hv*8 + r; all lanes converge, col==0 writes.
#pragma unroll
    for (int r = 0; r < 8; ++r) {
        float d  = bestD[r];
        int   ci = bestI[r];
#pragma unroll
        for (int m = 1; m < 16; m <<= 1) {
            float od = __shfl_xor(d,  m, 32);
            int   oi = __shfl_xor(ci, m, 32);
            if (od < d || (od == d && oi < ci)) { d = od; ci = oi; }
        }
        if (col == 0) {
            minidx[tokBase + hv * 8 + r] = ci;
            atomicAdd(&counts[ci], 1u);
        }
    }
}

// ========================================================================
// Kernel 2: noise substitution.  One block per token, 256 threads (one per
// dim), wave32 shuffle + LDS cross-wave reduction for the two norms.
// ========================================================================
__global__ __launch_bounds__(256)
void nsvq_finalize(const float* __restrict__ X,
                   const float* __restrict__ R,
                   const float* __restrict__ CB,
                   const int* __restrict__ minidx,
                   float* __restrict__ out)
{
    __shared__ float sR[8], sN[8];
    __shared__ float sScale;
    const int t    = blockIdx.x;
    const int d    = threadIdx.x;
    const int wave = d >> 5, lane = d & 31;

    const float x  = X[(size_t)t * EMB_DIM + d];
    const float rn = R[(size_t)t * EMB_DIM + d];
    const int idx  = minidx[t];
    const float q  = CB[(size_t)idx * EMB_DIM + d];

    const float dr = x - q;
    float sr = dr * dr;
    float sn = rn * rn;
#pragma unroll
    for (int m = 16; m >= 1; m >>= 1) {
        sr += __shfl_xor(sr, m, 32);
        sn += __shfl_xor(sn, m, 32);
    }
    if (lane == 0) { sR[wave] = sr; sN[wave] = sn; }
    __syncthreads();
    if (d == 0) {
        float tr = 0.0f, tn = 0.0f;
#pragma unroll
        for (int i = 0; i < 8; ++i) { tr += sR[i]; tn += sN[i]; }
        sScale = sqrtf(tr) / (sqrtf(tn) + NSVQ_EPS);
    }
    __syncthreads();
    out[(size_t)t * EMB_DIM + d] = x + sScale * rn;
}

// ========================================================================
// Kernel 3: perplexity + unique-code count from the usage counters.
// Single block, two-phase reduction (need total before entropy).
// ========================================================================
__global__ __launch_bounds__(1024)
void nsvq_perplexity(const unsigned* __restrict__ counts,
                     float* __restrict__ out)
{
    __shared__ float sT[32], sH[32], sU[32];
    __shared__ float sTotal;
    const int tid  = threadIdx.x;
    const int wave = tid >> 5, lane = tid & 31;

    float tot = 0.0f;
#pragma unroll
    for (int i = 0; i < NUM_EMB / 1024; ++i)
        tot += (float)counts[tid + i * 1024];
#pragma unroll
    for (int m = 16; m >= 1; m >>= 1) tot += __shfl_xor(tot, m, 32);
    if (lane == 0) sT[wave] = tot;
    __syncthreads();
    if (tid == 0) {
        float s = 0.0f;
#pragma unroll
        for (int i = 0; i < 32; ++i) s += sT[i];
        sTotal = fmaxf(s, 1.0f);
    }
    __syncthreads();

    const float inv = 1.0f / sTotal;
    float h = 0.0f, u = 0.0f;
#pragma unroll
    for (int i = 0; i < NUM_EMB / 1024; ++i) {
        const float c = (float)counts[tid + i * 1024];
        const float p = c * inv;
        h += p * logf(p + NSVQ_EPS);
        u += (c > 0.0f) ? 1.0f : 0.0f;
    }
#pragma unroll
    for (int m = 16; m >= 1; m >>= 1) {
        h += __shfl_xor(h, m, 32);
        u += __shfl_xor(u, m, 32);
    }
    if (lane == 0) { sH[wave] = h; sU[wave] = u; }
    __syncthreads();
    if (tid == 0) {
        float H = 0.0f, U = 0.0f;
#pragma unroll
        for (int i = 0; i < 32; ++i) { H += sH[i]; U += sU[i]; }
        out[(size_t)N_TOKENS * EMB_DIM]     = expf(-H);
        out[(size_t)N_TOKENS * EMB_DIM + 1] = U;
    }
}

// ========================================================================
extern "C" void kernel_launch(void* const* d_in, const int* in_sizes, int n_in,
                              void* d_out, int out_size, void* d_ws, size_t ws_size,
                              hipStream_t stream)
{
    (void)in_sizes; (void)n_in; (void)out_size; (void)ws_size;

    const float* X  = (const float*)d_in[0];   // input_data (N, D) f32
    const float* R  = (const float*)d_in[1];   // rand       (N, D) f32
    const float* CB = (const float*)d_in[2];   // codebooks  (K, D) f32
    float* out = (float*)d_out;                // [N*D quantized][perplexity][unique]

    // workspace layout (~192 KB)
    unsigned* counts = (unsigned*)d_ws;                 // K u32
    float*    cnorm  = (float*)d_ws + NUM_EMB;          // K f32
    int*      minidx = (int*)d_ws + 2 * NUM_EMB;        // N i32

    nsvq_prep      <<<NUM_EMB / 256, 256, 0, stream>>>(CB, cnorm, counts);
    nsvq_argmin    <<<N_TOKENS / BM, 256, 0, stream>>>(X, CB, cnorm, minidx, counts);
    nsvq_finalize  <<<N_TOKENS, 256, 0, stream>>>(X, R, CB, minidx, out);
    nsvq_perplexity<<<1, 1024, 0, stream>>>(counts, out);
}
